// Spatial_Stage_79156247265299
// MI455X (gfx1250) — matrix operand order repaired and checked by reading, not compiled, not run on hardware
//
#include <hip/hip_runtime.h>
#include <math.h>

// ---------------------------------------------------------------------------
// CDNA5 (gfx1250) implementation. All large channel-mixing ops lowered to
// v_wmma_f32_16x16x32_bf16 (wave32, 32x32 block per wave = 2x2 tiles, fp32
// accum, bf16 operands via v_cvt_pk_bf16_f32). WMMA path requires K % 32 == 0
// and N % 32 == 0 (true for every call); tiny-K layers (K=3/8) use a direct
// conv kernel instead. pha branch of the reference is dead code and dropped.
// ---------------------------------------------------------------------------

typedef __attribute__((ext_vector_type(16))) __bf16 v16bf;
typedef __attribute__((ext_vector_type(8)))  float  v8f;

#define GEMM_LRELU 1
#define GEMM_ACC   2

// A fragment (16x32 bf16): 16 contiguous K floats per lane.
static __device__ __forceinline__ v16bf load_a_frag(const float* __restrict__ pa) {
  const float4 a0 = *(const float4*)(pa);
  const float4 a1 = *(const float4*)(pa + 4);
  const float4 a2 = *(const float4*)(pa + 8);
  const float4 a3 = *(const float4*)(pa + 12);
  v16bf a;
  a[0]  = (__bf16)a0.x; a[1]  = (__bf16)a0.y; a[2]  = (__bf16)a0.z; a[3]  = (__bf16)a0.w;
  a[4]  = (__bf16)a1.x; a[5]  = (__bf16)a1.y; a[6]  = (__bf16)a1.z; a[7]  = (__bf16)a1.w;
  a[8]  = (__bf16)a2.x; a[9]  = (__bf16)a2.y; a[10] = (__bf16)a2.z; a[11] = (__bf16)a2.w;
  a[12] = (__bf16)a3.x; a[13] = (__bf16)a3.y; a[14] = (__bf16)a3.z; a[15] = (__bf16)a3.w;
  return a;
}

// B fragment (32x16 bf16): 16 consecutive K rows (stride N), one col per lane.
static __device__ __forceinline__ v16bf load_b_frag(const float* __restrict__ pb, size_t N) {
  v16bf b;
#pragma unroll
  for (int v = 0; v < 8; ++v) {
    b[2*v]   = (__bf16)pb[0];
    b[2*v+1] = (__bf16)pb[N];
    pb += 2 * N;
  }
  return b;
}

static __device__ __forceinline__ void store_tile(
    v8f acc, float* out, const float* __restrict__ bias,
    const float* __restrict__ mulp, const float* addp,
    int M, int N, int flags, int mbase, int n)
{
#pragma unroll
  for (int r = 0; r < 8; ++r) {
    int m = mbase + r;
    if (m < M) {
      float v = acc[r];
      if (bias) v += bias[m];
      if (flags & GEMM_LRELU) v = (v >= 0.f) ? v : 0.1f * v;
      size_t o = (size_t)m * N + n;
      if (mulp) v *= mulp[o];
      if (addp) v += addp[o];
      if (flags & GEMM_ACC) v += out[o];
      out[o] = v;
    }
  }
}

// D[M,N] = W[M,K] * X[K,N] (+bias)(lrelu?)(*mul)(+add)(+=out?)
// W row-major, row stride ldw. X row-major (K rows of N).
// REQUIRES: K % 32 == 0, N % 32 == 0. Each wave computes a 32x32 block.
__global__ __launch_bounds__(128) void wmma_gemm(
    const float* __restrict__ Wt, int ldw,
    const float* __restrict__ X,
    const float* __restrict__ bias,
    float* out,
    const float* __restrict__ mulp,
    const float* addp,
    int M, int N, int K, int flags)
{
  const int lane = threadIdx.x & 31;
  const int wave = threadIdx.x >> 5;
  const int tilesN = N >> 5;
  const int tilesM = (M + 31) >> 5;
  long t = (long)blockIdx.x * 4 + wave;
  if (t >= (long)tilesM * tilesN) return;       // wave-uniform exit (EXEC stays full)
  const int tm = (int)(t / tilesN);
  const int tn = (int)(t % tilesN);
  const int half = lane >> 4;                   // lane 0-15 / 16-31
  const int mn   = lane & 15;
  const int am0  = tm * 32 + mn;
  const int am1  = am0 + 16;
  const int am0c = (am0 < M) ? am0 : (M - 1);   // clamped: rows >= M never stored
  const int am1c = (am1 < M) ? am1 : (M - 1);
  const int bn0  = tn * 32 + mn;

  const float* __restrict__ ar0 = Wt + (size_t)am0c * ldw + (half << 4);
  const float* __restrict__ ar1 = Wt + (size_t)am1c * ldw + (half << 4);
  const float* __restrict__ xc0 = X + bn0;
  const float* __restrict__ xc1 = X + bn0 + 16;

  v8f c00 = {0.f,0.f,0.f,0.f,0.f,0.f,0.f,0.f};
  v8f c01 = c00, c10 = c00, c11 = c00;

  for (int k0 = 0; k0 < K; k0 += 32) {
    const v16bf a0 = load_a_frag(ar0 + k0);
    const v16bf a1 = load_a_frag(ar1 + k0);
    const size_t kb = (size_t)(k0 + (half << 4)) * N;
    const v16bf b0 = load_b_frag(xc0 + kb, (size_t)N);
    const v16bf b1 = load_b_frag(xc1 + kb, (size_t)N);
    c00 = __builtin_amdgcn_wmma_f32_16x16x32_bf16(false, a0, false, b0, (short)0, c00, false, false);
    c01 = __builtin_amdgcn_wmma_f32_16x16x32_bf16(false, a0, false, b1, (short)0, c01, false, false);
    c10 = __builtin_amdgcn_wmma_f32_16x16x32_bf16(false, a1, false, b0, (short)0, c10, false, false);
    c11 = __builtin_amdgcn_wmma_f32_16x16x32_bf16(false, a1, false, b1, (short)0, c11, false, false);
  }

  // C/D layout: lane holds col n, rows mbase..mbase+7 (mbase includes half*8)
  const int mb0 = tm * 32 + (half << 3);
  store_tile(c00, out, bias, mulp, addp, M, N, flags, mb0,      bn0);
  store_tile(c01, out, bias, mulp, addp, M, N, flags, mb0,      bn0 + 16);
  store_tile(c10, out, bias, mulp, addp, M, N, flags, mb0 + 16, bn0);
  store_tile(c11, out, bias, mulp, addp, M, N, flags, mb0 + 16, bn0 + 16);
}

// ---------------- tiny-K direct conv (K=3 / K=8 layers) ----------------
__global__ void k_conv1x1_small(const float* __restrict__ Wt, int K,
                                const float* __restrict__ X,
                                const float* __restrict__ bias,
                                float* __restrict__ out,
                                const float* __restrict__ addp,
                                int M, long N)
{
  long i = (long)blockIdx.x * blockDim.x + threadIdx.x;
  if (i >= (long)M * N) return;
  int m = (int)(i / N); long n = i % N;
  float acc = bias ? bias[m] : 0.f;
  for (int k = 0; k < K; ++k)
    acc += Wt[m * K + k] * X[(size_t)k * N + n];
  if (addp) acc += addp[i];
  out[i] = acc;
}

// ---------------- dark channel ----------------
__global__ void k_neg_min_c(const float* __restrict__ x, float* __restrict__ dm, long HW) {
  long i = (long)blockIdx.x * blockDim.x + threadIdx.x;
  if (i >= HW) return;
  float a = 1.f - x[i], b = 1.f - x[HW + i], c = 1.f - x[2*HW + i];
  dm[i] = fminf(a, fminf(b, c));
}
__global__ void k_hmin_reflect(const float* __restrict__ in, float* __restrict__ out, int H, int W) {
  long i = (long)blockIdx.x * blockDim.x + threadIdx.x;
  if (i >= (long)H * W) return;
  int w = (int)(i % W); long row = (i / W) * W;
  float m = 3.0e38f;
  for (int d = -7; d <= 7; ++d) {
    int ww = w + d;
    ww = ww < 0 ? -ww : (ww >= W ? 2*W - 2 - ww : ww);
    m = fminf(m, in[row + ww]);
  }
  out[i] = m;
}
__global__ void k_vmin_reflect(const float* __restrict__ in, float* __restrict__ out, int H, int W) {
  long i = (long)blockIdx.x * blockDim.x + threadIdx.x;
  if (i >= (long)H * W) return;
  int w = (int)(i % W); int h = (int)(i / W);
  float m = 3.0e38f;
  for (int d = -7; d <= 7; ++d) {
    int hh = h + d;
    hh = hh < 0 ? -hh : (hh >= H ? 2*H - 2 - hh : hh);
    m = fminf(m, in[(long)hh * W + w]);
  }
  out[i] = m;
}

// ---------------- max reduce (for ze normalization) ----------------
__global__ void k_zero_u32(unsigned* p) { if (threadIdx.x == 0 && blockIdx.x == 0) *p = 0u; }
__global__ void k_max_reduce(const float* __restrict__ d, long n, unsigned* out) {
  __shared__ float s[256];
  float m = 0.f;
  for (long i = (long)blockIdx.x * blockDim.x + threadIdx.x; i < n;
       i += (long)gridDim.x * blockDim.x)
    m = fmaxf(m, d[i]);
  s[threadIdx.x] = m; __syncthreads();
  for (int off = 128; off > 0; off >>= 1) {
    if ((int)threadIdx.x < off) s[threadIdx.x] = fmaxf(s[threadIdx.x], s[threadIdx.x + off]);
    __syncthreads();
  }
  if (threadIdx.x == 0) atomicMax(out, __float_as_uint(s[0]));   // vals >= 0
}

// ---------------- positional encoding (64 channels) ----------------
__global__ void k_posenc(const float* __restrict__ dark, const unsigned* __restrict__ zbits,
                         float* __restrict__ pos, int H, int W) {
  long HW = (long)H * W;
  long i = (long)blockIdx.x * blockDim.x + threadIdx.x;
  if (i >= HW) return;
  int w = (int)(i % W); int h = (int)(i / W);
  const float scale = 6.283185307179586f;
  float xe = (float)w / ((float)(W - 1) + 1e-6f) * scale;
  float ye = (float)h / ((float)(H - 1) + 1e-6f) * scale;
  float zm = __uint_as_float(*zbits);
  float ze = dark[i] / (zm + 1e-6f) * scale;
#pragma unroll
  for (int j = 0; j < 16; ++j) {
    float idt = __powf(10000.f, -(float)(j >> 1) * (1.f / 16.f));
    float px = xe * idt, py = ye * idt;
    pos[(long)j * HW + i]        = (j & 1) ? cosf(px) : sinf(px);
    pos[(long)(16 + j) * HW + i] = (j & 1) ? cosf(py) : sinf(py);
  }
#pragma unroll
  for (int j = 0; j < 32; ++j) {
    float idt = __powf(10000.f, -(float)(j >> 1) * (1.f / 16.f));
    float pz = ze * idt;
    pos[(long)(32 + j) * HW + i] = (j & 1) ? cosf(pz) : sinf(pz);
  }
}

// ---------------- space-to-depth (for 2x2 stride-2 conv as GEMM) ----------------
__global__ void k_s2d(const float* __restrict__ x, float* __restrict__ y,
                      int C, int Ho, int Wo) {
  long tot = (long)C * 4 * Ho * Wo;
  long i = (long)blockIdx.x * blockDim.x + threadIdx.x;
  if (i >= tot) return;
  long hw = i % ((long)Ho * Wo); int co = (int)(i / ((long)Ho * Wo));
  int wo = (int)(hw % Wo); int ho = (int)(hw / Wo);
  int c = co >> 2, kh = (co >> 1) & 1, kw = co & 1;
  y[i] = x[((long)c * (2 * Ho) + 2 * ho + kh) * (long)(2 * Wo) + 2 * wo + kw];
}

// ---------------- bilinear 2x upsample (half-pixel centers) ----------------
__global__ void k_up2(const float* __restrict__ in, float* __restrict__ out,
                      int C, int Hi, int Wi) {
  int Ho = Hi * 2, Wo = Wi * 2;
  long tot = (long)C * Ho * Wo;
  long i = (long)blockIdx.x * blockDim.x + threadIdx.x;
  if (i >= tot) return;
  int wo = (int)(i % Wo); long r = i / Wo; int ho = (int)(r % Ho); int c = (int)(r / Ho);
  float sy = (ho + 0.5f) * 0.5f - 0.5f;
  float sx = (wo + 0.5f) * 0.5f - 0.5f;
  int y0 = (int)floorf(sy); float fy = sy - (float)y0;
  int x0 = (int)floorf(sx); float fx = sx - (float)x0;
  int y1 = y0 + 1; if (y1 > Hi - 1) y1 = Hi - 1; if (y0 < 0) y0 = 0;
  int x1 = x0 + 1; if (x1 > Wi - 1) x1 = Wi - 1; if (x0 < 0) x0 = 0;
  const float* p = in + (long)c * Hi * Wi;
  float v00 = p[(long)y0 * Wi + x0], v01 = p[(long)y0 * Wi + x1];
  float v10 = p[(long)y1 * Wi + x0], v11 = p[(long)y1 * Wi + x1];
  out[i] = (1.f - fy) * ((1.f - fx) * v00 + fx * v01)
         +        fy  * ((1.f - fx) * v10 + fx * v11);
}

// ---------------- naive DFT kernels (incremental twiddles) ----------------
__global__ void k_rdft_w(const float* __restrict__ x, float* __restrict__ fre,
                         float* __restrict__ fim, int C, int H, int W, int F) {
  long tot = (long)C * H * F;
  long i = (long)blockIdx.x * blockDim.x + threadIdx.x;
  if (i >= tot) return;
  int f = (int)(i % F); long ch = i / F;
  const float* row = x + ch * W;
  float step = -6.283185307179586f * (float)f / (float)W;
  float cs = cosf(step), sn = sinf(step);
  float cr = 1.f, ci = 0.f, sre = 0.f, sim = 0.f;
  for (int w = 0; w < W; ++w) {
    float v = row[w];
    sre += v * cr; sim += v * ci;
    float nr = cr * cs - ci * sn; ci = cr * sn + ci * cs; cr = nr;
  }
  fre[i] = sre; fim[i] = sim;
}

__global__ void k_dft_h_abs(const float* __restrict__ fre, const float* __restrict__ fim,
                            float* __restrict__ amp, int C, int H, int F) {
  long tot = (long)C * H * F;
  long i = (long)blockIdx.x * blockDim.x + threadIdx.x;
  if (i >= tot) return;
  int f = (int)(i % F); long r = i / F; int k = (int)(r % H); int c = (int)(r / H);
  float step = -6.283185307179586f * (float)k / (float)H;
  float cs = cosf(step), sn = sinf(step);
  float cr = 1.f, ci = 0.f, are = 0.f, aim = 0.f;
  long base = ((long)c * H) * F + f;
  for (int h = 0; h < H; ++h) {
    float a = fre[base + (long)h * F], b = fim[base + (long)h * F];
    are += a * cr - b * ci; aim += a * ci + b * cr;
    float nr = cr * cs - ci * sn; ci = cr * sn + ci * cs; cr = nr;
  }
  amp[i] = sqrtf(are * are + aim * aim);
}

// z = amp * exp(i*amp)
__global__ void k_make_z(const float* __restrict__ amp, float* __restrict__ zre,
                         float* __restrict__ zim, long n) {
  long i = (long)blockIdx.x * blockDim.x + threadIdx.x;
  if (i >= n) return;
  float a = amp[i];
  zre[i] = a * cosf(a); zim[i] = a * sinf(a);
}

__global__ void k_idft_h(const float* __restrict__ zre, const float* __restrict__ zim,
                         float* __restrict__ yre, float* __restrict__ yim,
                         int C, int H, int F) {
  long tot = (long)C * H * F;
  long i = (long)blockIdx.x * blockDim.x + threadIdx.x;
  if (i >= tot) return;
  int f = (int)(i % F); long r = i / F; int h = (int)(r % H); int c = (int)(r / H);
  float step = 6.283185307179586f * (float)h / (float)H;
  float cs = cosf(step), sn = sinf(step);
  float cr = 1.f, ci = 0.f, sre = 0.f, sim = 0.f;
  long base = ((long)c * H) * F + f;
  for (int k = 0; k < H; ++k) {
    float a = zre[base + (long)k * F], b = zim[base + (long)k * F];
    sre += a * cr - b * ci; sim += a * ci + b * cr;
    float nr = cr * cs - ci * sn; ci = cr * sn + ci * cs; cr = nr;
  }
  float inv = 1.f / (float)H;
  yre[i] = sre * inv; yim[i] = sim * inv;
}

__global__ void k_irdft_w(const float* __restrict__ yre, const float* __restrict__ yim,
                          float* __restrict__ out, int C, int H, int W, int F) {
  long tot = (long)C * H * W;
  long i = (long)blockIdx.x * blockDim.x + threadIdx.x;
  if (i >= tot) return;
  int w = (int)(i % W); long r = i / W; int h = (int)(r % H); int c = (int)(r / H);
  float step = 6.283185307179586f * (float)w / (float)W;
  float cs = cosf(step), sn = sinf(step);
  long base = ((long)c * H + h) * F;
  float s = yre[base];                               // f = 0
  float cr = 1.f, ci = 0.f;
  for (int f = 1; f <= F - 2; ++f) {
    float nr = cr * cs - ci * sn; ci = cr * sn + ci * cs; cr = nr;
    s += 2.f * (yre[base + f] * cr - yim[base + f] * ci);
  }
  s += yre[base + F - 1] * ((w & 1) ? -1.f : 1.f);   // f = W/2 (Nyquist)
  out[i] = s / (float)W;
}

// ---------------------------------------------------------------------------
extern "C" void kernel_launch(void* const* d_in, const int* in_sizes, int n_in,
                              void* d_out, int out_size, void* d_ws, size_t ws_size,
                              hipStream_t stream) {
  (void)in_sizes; (void)n_in; (void)out_size; (void)ws_size;
  const float* x      = (const float*)d_in[0];
  const float* w_cf1  = (const float*)d_in[1];  const float* b_cf1 = (const float*)d_in[2];
  const float* w_e1   = (const float*)d_in[3];  const float* b_e1  = (const float*)d_in[4];
  const float* w_e2   = (const float*)d_in[5];  const float* b_e2  = (const float*)d_in[6];
  const float* w_d1   = (const float*)d_in[7];  const float* b_d1  = (const float*)d_in[8];
  const float* w_d2   = (const float*)d_in[9];  const float* b_d2  = (const float*)d_in[10];
  // pha branch (d_in[11..14]) is dead code in the reference: x_pha never feeds the output.
  const float* w_amp1 = (const float*)d_in[15]; const float* b_amp1= (const float*)d_in[16];
  const float* w_amp2 = (const float*)d_in[17]; const float* b_amp2= (const float*)d_in[18];
  const float* w_sp1  = (const float*)d_in[19]; const float* b_sp1 = (const float*)d_in[20];
  const float* w_sp2  = (const float*)d_in[21]; const float* b_sp2 = (const float*)d_in[22];
  const float* w_u1   = (const float*)d_in[23]; const float* b_u1  = (const float*)d_in[24];
  const float* w_u2   = (const float*)d_in[25]; const float* b_u2  = (const float*)d_in[26];
  const float* w_u3   = (const float*)d_in[27]; const float* b_u3  = (const float*)d_in[28];
  const float* w_u4   = (const float*)d_in[29]; const float* b_u4  = (const float*)d_in[30];
  const float* w_cf2  = (const float*)d_in[31]; const float* b_cf2 = (const float*)d_in[32];
  const float* w_cf3  = (const float*)d_in[33]; const float* b_cf3 = (const float*)d_in[34];
  float* outp = (float*)d_out;

  const int H = 512, W = 512;           // (1,3,512,512)
  const long HW  = (long)H * W;         // 262144
  const int H2 = 256, W2 = 256; const long HW2 = (long)H2 * W2;
  const int H4 = 128, W4 = 128; const long HW4 = (long)H4 * W4;
  const int F  = 65;  const long HF = (long)H4 * F;   // 8320 (= 32 * 260)

  float* ws = (float*)d_ws;
  // static workspace layout (floats), with reuse
  const size_t o_dm   = 0;
  const size_t o_dm2  = o_dm  + HW;
  const size_t o_dark = o_dm2 + HW;
  const size_t o_zmax = o_dark + HW;                // 1 u32 (padded 32)
  const size_t o_ape  = o_zmax + 32;                // 64*HW
  const size_t o_buf1 = o_ape  + 64 * HW;           // 16.78M (x_f / four1 / s2b / x_spat)
  const size_t o_buf2 = o_buf1 + 64 * HW;           // 16.78M (e1out / r1 / x_four)
  const size_t o_buf3 = o_buf2 + 64 * HW;           // 8.39M  (x_d1 / fourU1 / sU3)
  const size_t o_buf4 = o_buf3 + 128 * HW2;         // 8.39M  (r2 / four0)
  const size_t o_xd2  = o_buf4 + 512 * HW4;         // 4.19M  (x_d2; also pos & cf2-tmp span)
  const size_t o_f1   = o_xd2  + 256 * HW4;         // FFT block, 7 x 2.13M
  const size_t o_f2   = o_f1   + 256 * HF;
  const size_t o_amp  = o_f2   + 256 * HF;
  const size_t o_t1   = o_amp  + 256 * HF;
  const size_t o_amp2 = o_t1   + 256 * HF;
  const size_t o_yre  = o_amp2 + 256 * HF;
  const size_t o_yim  = o_yre  + 256 * HF;
  const size_t o_big  = o_yim  + 256 * HF;          // 33.55M (128ch @ 512x512, reused)
  const size_t o_pos  = o_xd2;                      // pos-enc (early) aliases xd2+fft span
  const size_t o_tmp  = o_xd2;                      // cf2 accumulator (late) same span
  const size_t o_s1   = o_f1;                       // spat intermediates reuse FFT block
  const size_t o_s2   = o_amp;

  auto blk = [](long n) { return dim3((unsigned)((n + 255) / 256)); };
  auto gemm = [&](const float* Wt, int ldw, const float* X, const float* bias,
                  float* out, const float* mulp, const float* addp,
                  int M, int N, int K, int flags) {
    long tiles = (long)((M + 31) / 32) * (N / 32);
    int blocks = (int)((tiles + 3) / 4);
    wmma_gemm<<<blocks, 128, 0, stream>>>(Wt, ldw, X, bias, out, mulp, addp, M, N, K, flags);
  };

  // ---- dark channel of (1 - x): min over c, then separable 15x15 min w/ reflect
  k_neg_min_c<<<blk(HW), 256, 0, stream>>>(x, ws + o_dm, HW);
  k_hmin_reflect<<<blk(HW), 256, 0, stream>>>(ws + o_dm,  ws + o_dm2, H, W);
  k_vmin_reflect<<<blk(HW), 256, 0, stream>>>(ws + o_dm2, ws + o_dark, H, W);

  // ---- max(dark) and positional encoding
  k_zero_u32<<<1, 32, 0, stream>>>((unsigned*)(ws + o_zmax));
  k_max_reduce<<<256, 256, 0, stream>>>(ws + o_dark, HW, (unsigned*)(ws + o_zmax));
  k_posenc<<<blk(HW), 256, 0, stream>>>(ws + o_dark, (const unsigned*)(ws + o_zmax),
                                        ws + o_pos, H, W);

  // ---- ape = pos_enc + conv1x1(x, cf1)   (K=3: direct conv)
  k_conv1x1_small<<<blk(64 * HW), 256, 0, stream>>>(w_cf1, 3, x, b_cf1,
                                                    ws + o_ape, ws + o_pos, 64, HW);

  // ---- x_f = e2(e1(x))   (K=3 and K=8: direct conv)
  k_conv1x1_small<<<blk(8 * HW), 256, 0, stream>>>(w_e1, 3, x, b_e1,
                                                   ws + o_buf2, nullptr, 8, HW);
  k_conv1x1_small<<<blk(64 * HW), 256, 0, stream>>>(w_e2, 8, ws + o_buf2, b_e2,
                                                    ws + o_buf1, nullptr, 64, HW);

  // ---- x_d1 = conv2x2s2(x_f)  via space-to-depth + WMMA GEMM
  k_s2d<<<blk(256 * HW2), 256, 0, stream>>>(ws + o_buf1, ws + o_buf2, 64, H2, W2);
  gemm(w_d1, 256, ws + o_buf2, b_d1, ws + o_buf3, nullptr, nullptr, 128, (int)HW2, 256, 0);

  // ---- x_d2 = conv2x2s2(x_d1)
  k_s2d<<<blk(512 * HW4), 256, 0, stream>>>(ws + o_buf3, ws + o_buf4, 128, H4, W4);
  gemm(w_d2, 512, ws + o_buf4, b_d2, ws + o_xd2, nullptr, nullptr, 256, (int)HW4, 512, 0);

  // ---- rfft2 -> |.|  (naive DFTs; pha branch dropped as dead code)
  k_rdft_w  <<<blk(256 * HF), 256, 0, stream>>>(ws + o_xd2, ws + o_f1, ws + o_f2, 256, H4, W4, F);
  k_dft_h_abs<<<blk(256 * HF), 256, 0, stream>>>(ws + o_f1, ws + o_f2, ws + o_amp, 256, H4, F);

  // ---- amp MLP
  gemm(w_amp1, 256, ws + o_amp, b_amp1, ws + o_t1,   nullptr, nullptr, 256, (int)HF, 256, GEMM_LRELU);
  gemm(w_amp2, 256, ws + o_t1,  b_amp2, ws + o_amp2, nullptr, nullptr, 256, (int)HF, 256, 0);

  // ---- irfft2(amp * e^{i amp})
  k_make_z <<<blk(256 * HF), 256, 0, stream>>>(ws + o_amp2, ws + o_f1, ws + o_f2, 256 * HF);
  k_idft_h <<<blk(256 * HF), 256, 0, stream>>>(ws + o_f1, ws + o_f2, ws + o_yre, ws + o_yim, 256, H4, F);
  k_irdft_w<<<blk(256 * HW4), 256, 0, stream>>>(ws + o_yre, ws + o_yim, ws + o_buf4, 256, H4, W4, F);

  // ---- four branch: up2 -> u1 -> up2 -> u2
  k_up2<<<blk(256 * HW2), 256, 0, stream>>>(ws + o_buf4, ws + o_buf1, 256, H4, W4);
  gemm(w_u1, 256, ws + o_buf1, b_u1, ws + o_buf3, nullptr, nullptr, 128, (int)HW2, 256, 0);
  k_up2<<<blk(128 * HW), 256, 0, stream>>>(ws + o_buf3, ws + o_big, 128, H2, W2);
  gemm(w_u2, 128, ws + o_big, b_u2, ws + o_buf2, nullptr, nullptr, 64, (int)HW, 128, 0);

  // ---- spat branch: MLP on x_d2 -> up2 -> u3 -> up2 -> u4 (* ape fused)
  gemm(w_sp1, 256, ws + o_xd2, b_sp1, ws + o_s1, nullptr, nullptr, 256, (int)HW4, 256, GEMM_LRELU);
  gemm(w_sp2, 256, ws + o_s1,  b_sp2, ws + o_s2, nullptr, nullptr, 256, (int)HW4, 256, 0);
  k_up2<<<blk(256 * HW2), 256, 0, stream>>>(ws + o_s2, ws + o_buf1, 256, H4, W4);
  gemm(w_u3, 256, ws + o_buf1, b_u3, ws + o_buf3, nullptr, nullptr, 128, (int)HW2, 256, 0);
  k_up2<<<blk(128 * HW), 256, 0, stream>>>(ws + o_buf3, ws + o_big, 128, H2, W2);
  gemm(w_u4, 128, ws + o_big, b_u4, ws + o_buf1, ws + o_ape, nullptr, 64, (int)HW, 128, 0);

  // ---- cf2 over concat(four, spat): split into two K=64 GEMMs (second accumulates)
  gemm(w_cf2,      128, ws + o_buf2, b_cf2,   ws + o_tmp, nullptr, nullptr, 64, (int)HW, 64, 0);
  gemm(w_cf2 + 64, 128, ws + o_buf1, nullptr, ws + o_tmp, nullptr, nullptr, 64, (int)HW, 64, GEMM_ACC);

  // ---- out = cf3(tmp) + x (residual fused)
  gemm(w_cf3, 64, ws + o_tmp, b_cf3, outp, nullptr, x, 3, (int)HW, 64, 0);
}
